// QualityRobustLoss_36232344109514
// MI455X (gfx1250) — matrix operand order repaired and checked by reading, not compiled, not run on hardware
//
#include <hip/hip_runtime.h>

// ---------------------------------------------------------------------------
// QualityRobustLoss for MI455X (gfx1250, wave32, WMMA)
//
// Pipeline (all on `stream`):
//   k_init    : zero/init accumulators in workspace
//   k_rownorm : L2-normalize feature rows, store bf16 fn[4096][512]
//   k_small   : focal / weighted-CE / boundary / evidential partials + class counts
//   k_gemm    : fn @ fn^T via v_wmma_f32_16x16x32_bf16 (2x4 register tiling,
//               1.5 b128-loads per WMMA), fused epilogue:
//                 pos / neg / semi sums, per-row hardest-pos max & hardest-neg min
//   k_final   : combine everything into the scalar loss
// ---------------------------------------------------------------------------

#define B_N 4096
#define D_K 512

typedef __attribute__((ext_vector_type(16))) __bf16 v16bf;
typedef __attribute__((ext_vector_type(8)))  __bf16 v8bf;
typedef __attribute__((ext_vector_type(2)))  __bf16 v2bf;
typedef __attribute__((ext_vector_type(8)))  float  v8f;

union ABu { v16bf v; v8bf h[2]; };

// ---- workspace layout (bytes) ----
#define WS_HP     0          // 4096 * 4  (uint bits of hardest-positive dist)
#define WS_HN     16384      // 4096 * 4  (uint bits of hardest-negative dist)
#define WS_CNT    32768      // 3 * 4     (class counts, int)
#define WS_SMALL  33024      // 16 blocks * 8 floats
#define WS_GEMM   34048      // 1024 blocks * 4 floats (pos, neg, semi, pad)
#define WS_FN     65536      // 4096*512*2 bytes bf16 normalized features

// ---------------------------------------------------------------------------
__device__ __forceinline__ float block_reduce_sum(float v, float* sbuf) {
  int tid = threadIdx.x;
  sbuf[tid] = v;
  __syncthreads();
  for (int s = blockDim.x >> 1; s > 0; s >>= 1) {
    if (tid < s) sbuf[tid] += sbuf[tid + s];
    __syncthreads();
  }
  float r = sbuf[0];
  __syncthreads();
  return r;
}

__device__ __forceinline__ float digammaf_dev(float x) {
  // recurrence to x >= 6, then asymptotic series
  float r = 0.f;
  while (x < 6.f) { r -= 1.f / x; x += 1.f; }
  float inv = 1.f / x, inv2 = inv * inv;
  return r + logf(x) - 0.5f * inv
         - inv2 * (1.f/12.f - inv2 * (1.f/120.f - inv2 * (1.f/252.f)));
}

// ---------------------------------------------------------------------------
__global__ void qrl_init(unsigned* hpB, unsigned* hnB, int* counts,
                         float* smallP, float* gemmP) {
  int idx = blockIdx.x * blockDim.x + threadIdx.x;
  int stride = gridDim.x * blockDim.x;
  for (int i = idx; i < B_N; i += stride) { hpB[i] = 0u; hnB[i] = 0x7F800000u; }
  for (int i = idx; i < 16 * 8; i += stride) smallP[i] = 0.f;
  for (int i = idx; i < 1024 * 4; i += stride) gemmP[i] = 0.f;
  if (idx < 3) counts[idx] = 0;
}

// ---------------------------------------------------------------------------
__global__ void qrl_rownorm(const float* __restrict__ feat,
                            unsigned short* __restrict__ fnraw) {
  __bf16* fn = (__bf16*)fnraw;
  int row = blockIdx.x;
  int tid = threadIdx.x;                       // 256 threads, 2 elems each
  const float* fr = feat + (size_t)row * D_K;
  float2 v = *(const float2*)(fr + tid * 2);
  float ss = v.x * v.x + v.y * v.y;
  __shared__ float sb[256];
  ss = block_reduce_sum(ss, sb);
  float inv = 1.f / fmaxf(sqrtf(ss), 1e-12f);
  v2bf p;
  p[0] = (__bf16)(v.x * inv);
  p[1] = (__bf16)(v.y * inv);
  *(v2bf*)(fn + (size_t)row * D_K + tid * 2) = p;
}

// ---------------------------------------------------------------------------
__global__ void qrl_small(const float* __restrict__ logits,
                          const int* __restrict__ targets,
                          const float* __restrict__ alpha,
                          const float* __restrict__ focal_alpha,
                          const float* __restrict__ class_weights,
                          float* __restrict__ smallP,
                          int* __restrict__ counts) {
  int i = blockIdx.x * blockDim.x + threadIdx.x;  // 16 blocks x 256 = 4096
  float l0 = logits[i*3+0], l1 = logits[i*3+1], l2 = logits[i*3+2];
  float m  = fmaxf(l0, fmaxf(l1, l2));
  float e0 = expf(l0-m), e1 = expf(l1-m), e2 = expf(l2-m);
  float se = e0+e1+e2;
  float lse = m + logf(se);
  float lp0 = l0-lse, lp1 = l1-lse, lp2 = l2-lse;
  int t = targets[i];
  float lpt = (t==0) ? lp0 : ((t==1) ? lp1 : lp2);
  float nll = -lpt;
  float ce_ls = 0.9f*nll + 0.1f*(-(lp0+lp1+lp2)*(1.f/3.f));
  float pt = expf(-ce_ls);
  float om = 1.f - pt;
  float focal = focal_alpha[t] * om*om * ce_ls;
  float w = class_weights[t];
  float wnll = w*nll;
  float inv_se = 1.f/se;
  float p0 = e0*inv_se, p1 = e1*inv_se, p2 = e2*inv_se;
  float rb = (t==0) ? (p1 + 0.6f*p2) : 0.f;
  float sbb = (t==2) ? (p1 + 0.4f*p0) : 0.f;
  float mb = (t==1) ? (1.f - p1 + 0.3f*(p0+p2)) : 0.f;

  // evidential
  float a0 = alpha[i*3+0], a1 = alpha[i*3+1], a2 = alpha[i*3+2];
  float S  = a0+a1+a2 + 1e-8f;
  float a_t = (t==0) ? a0 : ((t==1) ? a1 : a2);
  float lik = digammaf_dev(S) - digammaf_dev(a_t + 1e-8f);
  float b0 = (t==0) ? 1.f : a0;
  float b1 = (t==1) ? 1.f : a1;
  float b2 = (t==2) ? 1.f : a2;
  float bs = b0+b1+b2 + 1e-8f;
  float dgs = digammaf_dev(bs + 1e-8f);
  float kl = lgammaf(bs)
           - (lgammaf(b0+1e-8f) + lgammaf(b1+1e-8f) + lgammaf(b2+1e-8f))
           + (b0-1.f)*(digammaf_dev(b0+1e-8f) - dgs)
           + (b1-1.f)*(digammaf_dev(b1+1e-8f) - dgs)
           + (b2-1.f)*(digammaf_dev(b2+1e-8f) - dgs);

  atomicAdd(&counts[t], 1);

  __shared__ float sb[256];
  float r;
  r = block_reduce_sum(focal, sb); if (threadIdx.x==0) smallP[blockIdx.x*8+0] = r;
  r = block_reduce_sum(wnll , sb); if (threadIdx.x==0) smallP[blockIdx.x*8+1] = r;
  r = block_reduce_sum(w    , sb); if (threadIdx.x==0) smallP[blockIdx.x*8+2] = r;
  r = block_reduce_sum(rb   , sb); if (threadIdx.x==0) smallP[blockIdx.x*8+3] = r;
  r = block_reduce_sum(sbb  , sb); if (threadIdx.x==0) smallP[blockIdx.x*8+4] = r;
  r = block_reduce_sum(mb   , sb); if (threadIdx.x==0) smallP[blockIdx.x*8+5] = r;
  r = block_reduce_sum(lik  , sb); if (threadIdx.x==0) smallP[blockIdx.x*8+6] = r;
  r = block_reduce_sum(kl   , sb); if (threadIdx.x==0) smallP[blockIdx.x*8+7] = r;
}

// ---------------------------------------------------------------------------
// 128x128 sim super-tile per block (8 waves).  Each wave computes a 32x64
// slab as a 2x4 grid of 16x16 WMMA tiles: 2 A-fragments x 4 B-fragments
// reuse both operands, so each k-step issues 12 b128 loads for 8 WMMAs
// (1.5 loads/WMMA vs 2.25 for a 1x8 tiling).  A/B fragments load straight
// from row-major bf16 memory in the exact per-lane layout
// v_wmma_f32_16x16x32_bf16 expects.
// ---------------------------------------------------------------------------
__global__ void __launch_bounds__(256)
qrl_gemm(const unsigned short* __restrict__ fnraw,
         const int* __restrict__ targets,
         unsigned* __restrict__ hpB,
         unsigned* __restrict__ hnB,
         float* __restrict__ gemmP) {
  const __bf16* fn = (const __bf16*)fnraw;
  const int lane = threadIdx.x & 31;
  const int wave = threadIdx.x >> 5;
  const int lane15 = lane & 15;
  const int hi = lane >> 4;

  // wave (w>>1) picks one of 4 32-row groups; (w&1) picks one of 2 64-col groups
  const int i0 = (blockIdx.x >> 5) * 128 + (wave >> 1) * 32;
  const int j0 = (blockIdx.x & 31) * 128 + (wave & 1) * 64;

  // A fragment bases: lane L holds row (i0[+16] + L%16), K = {0..7,16..23} (+8 if hi)
  const __bf16* arow0 = fn + (size_t)(i0 + lane15) * D_K + hi * 8;
  const __bf16* arow1 = arow0 + (size_t)16 * D_K;
  // B fragment bases: lane L holds col (j0+t*16 + L%16), K = {0..7,16..23} (+8 if hi)
  const __bf16* brow[4];
#pragma unroll
  for (int t = 0; t < 4; ++t)
    brow[t] = fn + (size_t)(j0 + t * 16 + lane15) * D_K + hi * 8;

  v8f acc[2][4] = {};
  for (int kt = 0; kt < D_K; kt += 32) {
    __builtin_prefetch(arow0 + kt + 64, 0, 1);    // global_prefetch_b8
    ABu a0, a1;
    a0.h[0] = *(const v8bf*)(arow0 + kt);
    a0.h[1] = *(const v8bf*)(arow0 + kt + 16);
    a1.h[0] = *(const v8bf*)(arow1 + kt);
    a1.h[1] = *(const v8bf*)(arow1 + kt + 16);
#pragma unroll
    for (int t = 0; t < 4; ++t) {
      ABu b;
      b.h[0] = *(const v8bf*)(brow[t] + kt);
      b.h[1] = *(const v8bf*)(brow[t] + kt + 16);
      acc[0][t] = __builtin_amdgcn_wmma_f32_16x16x32_bf16(
          false, a0.v, false, b.v, (short)0, acc[0][t], false, false);
      acc[1][t] = __builtin_amdgcn_wmma_f32_16x16x32_bf16(
          false, a1.v, false, b.v, (short)0, acc[1][t], false, false);
    }
  }

  // ---- fused epilogue: reductions over the 32x64 slab ----
  float pos = 0.f, neg = 0.f, semi = 0.f;
  float hp[2][8], hn[2][8];
#pragma unroll
  for (int r = 0; r < 2; ++r)
#pragma unroll
    for (int v = 0; v < 8; ++v) { hp[r][v] = 0.f; hn[r][v] = __builtin_inff(); }

  int tiv[2][8];
#pragma unroll
  for (int r = 0; r < 2; ++r)
#pragma unroll
    for (int v = 0; v < 8; ++v) tiv[r][v] = targets[i0 + r * 16 + v + hi * 8];

#pragma unroll
  for (int t = 0; t < 4; ++t) {
    int j  = j0 + t * 16 + lane15;
    int tj = targets[j];
#pragma unroll
    for (int r = 0; r < 2; ++r) {
#pragma unroll
      for (int v = 0; v < 8; ++v) {
        int i = i0 + r * 16 + v + hi * 8;
        if (i == j) continue;                      // diag: all terms vanish
        float s  = acc[r][t][v];
        float d1 = 1.f - s;
        float d2 = fmaxf(2.f - 2.f * s, 0.f);
        float dist = sqrtf(d2);
        if (tiv[r][v] == tj) {
          pos += d1 * d1;
          if (tiv[r][v] == 1) semi += d1 * d1;
          hp[r][v] = fmaxf(hp[r][v], dist);
        } else {
          float c = fmaxf(s - 0.2f, 0.f);
          neg += c * c;
          hn[r][v] = fminf(hn[r][v], dist);
        }
      }
    }
  }

  // per-row reduce across the 16 lanes of each half, then atomic combine
#pragma unroll
  for (int r = 0; r < 2; ++r) {
#pragma unroll
    for (int v = 0; v < 8; ++v) {
      float a = hp[r][v], b = hn[r][v];
      for (int msk = 8; msk >= 1; msk >>= 1) {
        a = fmaxf(a, __shfl_xor(a, msk, 32));
        b = fminf(b, __shfl_xor(b, msk, 32));
      }
      if (lane15 == 0) {
        int row = i0 + r * 16 + v + hi * 8;
        atomicMax(&hpB[row], __float_as_uint(a));  // dist >= 0: uint order OK
        atomicMin(&hnB[row], __float_as_uint(b));
      }
    }
  }

  // scalar wave reduce, then fixed-order block combine (deterministic)
  for (int msk = 16; msk >= 1; msk >>= 1) {
    pos  += __shfl_xor(pos , msk, 32);
    neg  += __shfl_xor(neg , msk, 32);
    semi += __shfl_xor(semi, msk, 32);
  }
  __shared__ float sred[3][8];
  if (lane == 0) { sred[0][wave] = pos; sred[1][wave] = neg; sred[2][wave] = semi; }
  __syncthreads();
  if (threadIdx.x == 0) {
    float p = 0.f, n = 0.f, sm = 0.f;
    for (int wv = 0; wv < 8; ++wv) { p += sred[0][wv]; n += sred[1][wv]; sm += sred[2][wv]; }
    gemmP[blockIdx.x * 4 + 0] = p;
    gemmP[blockIdx.x * 4 + 1] = n;
    gemmP[blockIdx.x * 4 + 2] = sm;
  }
}

// ---------------------------------------------------------------------------
__global__ void qrl_final(const int* __restrict__ targets,
                          const int* __restrict__ epochp,
                          const unsigned* __restrict__ hpB,
                          const unsigned* __restrict__ hnB,
                          const int* __restrict__ counts,
                          const float* __restrict__ smallP,
                          const float* __restrict__ gemmP,
                          float* __restrict__ out) {
  __shared__ float sb[256];
  int tid = threadIdx.x;
  int c0 = counts[0], c1 = counts[1], c2 = counts[2];

  float pos = 0.f, neg = 0.f, semi = 0.f;
  for (int k = tid; k < 1024; k += 256) {
    pos  += gemmP[k * 4 + 0];
    neg  += gemmP[k * 4 + 1];
    semi += gemmP[k * 4 + 2];
  }
  pos  = block_reduce_sum(pos , sb);
  neg  = block_reduce_sum(neg , sb);
  semi = block_reduce_sum(semi, sb);

  float tls = 0.f, cnt = 0.f;
  for (int i = tid; i < B_N; i += 256) {
    int t  = targets[i];
    int nc = (t == 0) ? c0 : ((t == 1) ? c1 : c2);
    if (nc >= 2 && (B_N - nc) >= 1) {
      float hp = __uint_as_float(hpB[i]);
      float hn = __uint_as_float(hnB[i]);
      float margin = 1.5f * ((t == 1) ? 2.5f : 1.f);
      tls += fmaxf(hp - hn + margin, 0.f);
      cnt += 1.f;
    }
  }
  tls = block_reduce_sum(tls, sb);
  cnt = block_reduce_sum(cnt, sb);

  if (tid == 0) {
    float fsum=0,wnll=0,wsum=0,rb=0,sbb=0,mb=0,lik=0,kl=0;
    for (int b = 0; b < 16; ++b) {
      fsum += smallP[b*8+0]; wnll += smallP[b*8+1]; wsum += smallP[b*8+2];
      rb   += smallP[b*8+3]; sbb  += smallP[b*8+4]; mb   += smallP[b*8+5];
      lik  += smallP[b*8+6]; kl   += smallP[b*8+7];
    }
    const float Bf = (float)B_N;
    float focal    = fsum / Bf;
    float ce       = wnll / wsum;
    float boundary = (rb + sbb + 4.5f * mb) / (Bf + 1e-8f);
    float total    = 0.4f * focal + 0.3f * ce + 0.15f * boundary;

    float contrastive = (pos + neg + 4.f * semi) / (Bf * Bf + 1e-8f);
    float triplet = (cnt > 0.f) ? (tls / fmaxf(cnt, 1.f)) : 0.f;

    double sq = (double)c0*c0 + (double)c1*c1 + (double)c2*c2;
    float npairs = (float)((sq - (double)B_N) * 0.5);
    float q = (pos * 0.5f) / fmaxf(npairs, 1.f);    // upper-tri same-class sum == pos/2
    if (c2 > 0) q *= (1.f + 2.5f * ((float)c2 / Bf));
    float quality = (npairs > 0.f) ? q : 0.f;

    float ep = (float)epochp[0] / 25.f;
    float ann = fminf(1.f, ep * ep);
    float evid = lik / Bf + ann * 0.2f * (kl / Bf);

    total += 0.1f * contrastive + 0.1f * triplet + 0.1f * quality + 0.1f * evid;
    out[0] = total;
  }
}

// ---------------------------------------------------------------------------
extern "C" void kernel_launch(void* const* d_in, const int* in_sizes, int n_in,
                              void* d_out, int out_size, void* d_ws, size_t ws_size,
                              hipStream_t stream) {
  const float* logits        = (const float*)d_in[0];
  const int*   targets       = (const int*)  d_in[1];
  const float* features      = (const float*)d_in[2];
  const float* alpha         = (const float*)d_in[3];
  const int*   epoch         = (const int*)  d_in[4];
  const float* focal_alpha   = (const float*)d_in[5];
  const float* class_weights = (const float*)d_in[6];
  float* out = (float*)d_out;

  char* ws = (char*)d_ws;                 // needs ~4.3 MB
  unsigned* hpB   = (unsigned*)(ws + WS_HP);
  unsigned* hnB   = (unsigned*)(ws + WS_HN);
  int*      cnts  = (int*)     (ws + WS_CNT);
  float*    smP   = (float*)   (ws + WS_SMALL);
  float*    gmP   = (float*)   (ws + WS_GEMM);
  unsigned short* fn = (unsigned short*)(ws + WS_FN);

  qrl_init<<<16, 256, 0, stream>>>(hpB, hnB, cnts, smP, gmP);
  qrl_rownorm<<<B_N, 256, 0, stream>>>(features, fn);
  qrl_small<<<B_N / 256, 256, 0, stream>>>(logits, targets, alpha,
                                           focal_alpha, class_weights, smP, cnts);
  qrl_gemm<<<1024, 256, 0, stream>>>(fn, targets, hpB, hnB, gmP);
  qrl_final<<<1, 256, 0, stream>>>(targets, epoch, hpB, hnB, cnts, smP, gmP, out);
}
